// LossModule_58213986730076
// MI455X (gfx1250) — matrix operand order repaired and checked by reading, not compiled, not run on hardware
//
#include <hip/hip_runtime.h>

// ---------------- problem constants ----------------
constexpr int Bq   = 8192;   // rows of v / vhat
constexpr int Dd   = 256;    // feature dim (contraction)
constexpr int Nn   = 2048;   // negatives
constexpr int Kk   = 512;    // rows of F
constexpr int Tt   = 8;      // top-k
constexpr float LAMBDA_ORTHO = 1e-3f;
constexpr float EPS          = 1e-10f;

constexpr int KB = Dd / 32;          // k-blocks of 32 (bf16 WMMA K)
constexpr int FRAG_ELEMS = 512;      // 16x32 bf16 fragment = 32 lanes * 16 elems

typedef __attribute__((ext_vector_type(16))) __bf16 v16bf;
typedef __attribute__((ext_vector_type(8)))  float  v8f;
typedef __attribute__((ext_vector_type(4)))  unsigned int v4u;
typedef __attribute__((ext_vector_type(4)))  int v4i;
typedef __attribute__((ext_vector_type(8)))  int v8i;

#if defined(__has_builtin)
#if __has_builtin(__builtin_amdgcn_tensor_load_to_lds) && \
    __has_builtin(__builtin_amdgcn_s_wait_tensorcnt)
#define USE_TDM 1
#endif
#endif
#ifndef USE_TDM
#define USE_TDM 0
#endif

__device__ __forceinline__ float wave_sum(float x) {
#pragma unroll
  for (int off = 16; off > 0; off >>= 1) x += __shfl_xor(x, off, 32);
  return x;
}

#if USE_TDM
// Issue a TDM 2D tile load: 8 rows x 1024B (one k-slice of 8 B-fragments),
// row stride KB*1024B, into contiguous LDS at lds_off.
// D# per CDNA5 ISA 8.3/8.4: data_size=4B, tensor_dim0=256, tensor_dim1=8,
// tile_dim0=256, tile_dim1=8, tensor_dim0_stride=2048 (4B units).
__device__ __forceinline__ void tdm_load_btile(const void* gbase,
                                               unsigned lds_off) {
  unsigned long long ga = (unsigned long long)gbase;
  v4u g0 = {1u,                               // count=1, user desc
            lds_off,                          // lds_addr (bytes)
            (unsigned)(ga & 0xFFFFFFFFu),     // global_addr[31:0]
            (unsigned)((ga >> 32) & 0x1FFFFFFu) | (2u << 30)};  // [56:32]|type=2
  v8i g1 = {(int)(2u << 16),     // wg_mask=0, data_size=2 (4 bytes)
            (int)(256u << 16),   // tensor_dim0[15:0]=256
            (int)(8u << 16),     // tensor_dim0[31:16]=0, tensor_dim1[15:0]=8
            (int)(256u << 16),   // tensor_dim1[31:16]=0, tile_dim0=256
            (int)8,              // tile_dim1=8, tile_dim2=0
            (int)2048,           // tensor_dim0_stride[31:0]=2048 (4B units)
            0, 0};               // stride hi / tensor_dim1_stride = 0
  v4i gz = {0, 0, 0, 0};
#if __clang_major__ >= 23
  v8i gz8 = {0, 0, 0, 0, 0, 0, 0, 0};
  __builtin_amdgcn_tensor_load_to_lds(g0, g1, gz, gz, gz8, 0);
#else
  __builtin_amdgcn_tensor_load_to_lds(g0, g1, gz, gz, 0);
#endif
}
#endif

// ---------------------------------------------------------------------------
// Pack f32 rows into WMMA A-matrix fragments (16x32 bf16), hi/lo split.
// A layout (16-bit A 16x32): lane half h = lane>>4, M = lane&15,
// element j:  K = 16*h + j  (contiguous, matches B packing)
// ---------------------------------------------------------------------------
__global__ void pack_a_kernel(const float* __restrict__ src,
                              __bf16* __restrict__ dhi,
                              __bf16* __restrict__ dlo, int rows) {
  int wid  = (blockIdx.x * blockDim.x + threadIdx.x) >> 5;
  int lane = threadIdx.x & 31;
  int nfrag = (rows / 16) * KB;
  if (wid >= nfrag) return;
  int mb = wid / KB, kb = wid % KB;
  int hf = lane >> 4;
  int m  = mb * 16 + (lane & 15);
  size_t base = (size_t)wid * FRAG_ELEMS + (size_t)lane * 16;
  const float* s = src + (size_t)m * Dd + (size_t)kb * 32;
#pragma unroll
  for (int j = 0; j < 16; ++j) {
    int v = j >> 1, sb = j & 1;
    int kk = 16 * hf + 2 * v + sb;
    float x = s[kk];
    __bf16 hi = (__bf16)x;
    float  r  = x - (float)hi;
    dhi[base + j] = hi;
    dlo[base + j] = (__bf16)r;
  }
}

// ---------------------------------------------------------------------------
// Pack f32 rows into WMMA B-matrix fragments (32x16 bf16, K x N), hi/lo split.
// B layout: lanes 0-15 hold K=0..15 (N=lane), lanes 16-31 hold K=16..31.
// Within a lane: K = kb*32 + 16*half + j  (j = 0..15, contiguous)
// ---------------------------------------------------------------------------
__global__ void pack_b_kernel(const float* __restrict__ src,
                              __bf16* __restrict__ dhi,
                              __bf16* __restrict__ dlo, int rows) {
  int wid  = (blockIdx.x * blockDim.x + threadIdx.x) >> 5;
  int lane = threadIdx.x & 31;
  int nfrag = (rows / 16) * KB;
  if (wid >= nfrag) return;
  int nb = wid / KB, kb = wid % KB;
  int hf = lane >> 4;
  int n  = nb * 16 + (lane & 15);
  size_t base = (size_t)wid * FRAG_ELEMS + (size_t)lane * 16;
  const float* s = src + (size_t)n * Dd + (size_t)kb * 32 + 16 * hf;
#pragma unroll
  for (int j = 0; j < 16; ++j) {
    float x = s[j];
    __bf16 hi = (__bf16)x;
    float  r  = x - (float)hi;
    dhi[base + j] = hi;
    dlo[base + j] = (__bf16)r;
  }
}

// ---------------------------------------------------------------------------
// Per-row stats for v/vhat: true_dist, c1 = 1 + td - ||vhat||^2, maskf.
// One wave per row.
// ---------------------------------------------------------------------------
__global__ void row_stats_kernel(const float* __restrict__ v,
                                 const float* __restrict__ vhat,
                                 const unsigned char* __restrict__ mask,
                                 float* __restrict__ c1,
                                 float* __restrict__ td,
                                 float* __restrict__ w) {
  int wid  = (blockIdx.x * blockDim.x + threadIdx.x) >> 5;
  int lane = threadIdx.x & 31;
  if (wid >= Bq) return;
  const float4* pv = (const float4*)(v    + (size_t)wid * Dd);
  const float4* ph = (const float4*)(vhat + (size_t)wid * Dd);
  float tdl = 0.f, vnl = 0.f;
#pragma unroll
  for (int i = 0; i < 2; ++i) {
    float4 a = ph[lane * 2 + i];
    float4 c = pv[lane * 2 + i];
    float d0 = a.x - c.x, d1 = a.y - c.y, d2 = a.z - c.z, d3 = a.w - c.w;
    tdl += d0 * d0 + d1 * d1 + d2 * d2 + d3 * d3;
    vnl += a.x * a.x + a.y * a.y + a.z * a.z + a.w * a.w;
  }
  float t  = wave_sum(tdl);
  float vn = wave_sum(vnl);
  if (lane == 0) {
    td[wid] = t;
    c1[wid] = 1.0f + t - vn;
    w[wid]  = mask[wid] ? 1.0f : 0.0f;
  }
}

// ||row||^2 for negatives. One wave per row.
__global__ void rownorm_kernel(const float* __restrict__ src,
                               float* __restrict__ out, int rows) {
  int wid  = (blockIdx.x * blockDim.x + threadIdx.x) >> 5;
  int lane = threadIdx.x & 31;
  if (wid >= rows) return;
  const float4* p = (const float4*)(src + (size_t)wid * Dd);
  float s = 0.f;
#pragma unroll
  for (int i = 0; i < 2; ++i) {
    float4 a = p[lane * 2 + i];
    s += a.x * a.x + a.y * a.y + a.z * a.z + a.w * a.w;
  }
  s = wave_sum(s);
  if (lane == 0) out[wid] = s;
}

// ---------------------------------------------------------------------------
// Fused GEMM + epilogue.  Block tile: 64(M) x 128(N), 8 waves (4Mx2N),
// each wave: 16x64 via 4 f32 accumulators.  3 WMMA per acc per k-step
// (hi*hi + hi*lo + lo*hi split-precision bf16).
// B tiles staged to LDS by the Tensor Data Mover (double-buffered,
// TENSORcnt-synchronized) when available; manual ds staging otherwise.
// EPI==0: contrastive hinge  sum w[row]*max(c1[row] + 2S - negnorm[col], 0)
// EPI==1: ortho              sum (S - (row==col))^2
// ---------------------------------------------------------------------------
template <int EPI>
__global__ __launch_bounds__(256) void gemm_kernel(
    const __bf16* __restrict__ Ahi, const __bf16* __restrict__ Alo,
    const __bf16* __restrict__ Bhi, const __bf16* __restrict__ Blo,
    const float* __restrict__ c1, const float* __restrict__ w,
    const float* __restrict__ negnorm, float* __restrict__ partials) {
  __shared__ __align__(32) unsigned char ldsB[32768];  // 2 x (8 nb hi+lo)
  __shared__ float wsums[8];

  const int tid  = threadIdx.x;
  const int wid  = tid >> 5;
  const int lane = tid & 31;
  const int hf   = lane >> 4;
  const int waveM = wid & 3, waveN = wid >> 2;
  const int blockM = blockIdx.y, blockN = blockIdx.x;
  const int mb = blockM * 4 + waveM;  // A fragment row-block (16 rows)

  v8f acc[4];
#pragma unroll
  for (int i = 0; i < 4; ++i)
#pragma unroll
    for (int r = 0; r < 8; ++r) acc[i][r] = 0.f;

#if USE_TDM
  const unsigned ldsBase = (unsigned)(uintptr_t)(void*)ldsB;
  // preload k-slice 0 into buffer 0 (one wave drives the DMA engine)
  if (wid == 0) {
    size_t g0 = ((size_t)blockN * 8 * KB + 0) * 1024;
    tdm_load_btile((const char*)Bhi + g0, ldsBase);
    tdm_load_btile((const char*)Blo + g0, ldsBase + 8192);
  }
#endif

  for (int kb = 0; kb < KB; ++kb) {
#if USE_TDM
    const unsigned bufOff = (kb & 1) ? 16384u : 0u;
    if (wid == 0) {
      if (kb + 1 < KB) {  // prefetch next slice into the other buffer
        size_t gn = ((size_t)blockN * 8 * KB + (kb + 1)) * 1024;
        unsigned nb = ((kb + 1) & 1) ? 16384u : 0u;
        tdm_load_btile((const char*)Bhi + gn, ldsBase + nb);
        tdm_load_btile((const char*)Blo + gn, ldsBase + nb + 8192);
        __builtin_amdgcn_s_wait_tensorcnt(2);  // current slice complete
      } else {
        __builtin_amdgcn_s_wait_tensorcnt(0);
      }
    }
    __syncthreads();
#else
    const unsigned bufOff = 0u;
    {  // ---- manual staging: 8 B fragments (hi+lo) for this k-slice ----
      const uint4* srcHi = (const uint4*)Bhi;
      const uint4* srcLo = (const uint4*)Blo;
      uint4* dst = (uint4*)ldsB;
#pragma unroll
      for (int c = 0; c < 4; ++c) {
        int i = tid * 4 + c;          // 0..1023 uint4s (16 KB)
        int isLo = i >> 9;
        int r2 = i & 511;
        int nbL = r2 >> 6, u = r2 & 63;
        size_t gidx = ((size_t)(blockN * 8 + nbL) * KB + kb) * 64 + u;
        dst[i] = isLo ? srcLo[gidx] : srcHi[gidx];
      }
    }
    __syncthreads();
    if (kb + 1 < KB) {  // prefetch next k-slice of B into L2
      size_t nxt = ((size_t)(blockN * 8 + wid) * KB + (kb + 1)) * 1024 +
                   (size_t)lane * 32;
      __builtin_prefetch((const char*)Bhi + nxt, 0, 0);
      __builtin_prefetch((const char*)Blo + nxt, 0, 0);
    }
#endif

    // ---- A fragments (direct from global, 32 B/lane each) ----
    size_t aoff = ((size_t)mb * KB + kb) * FRAG_ELEMS + (size_t)lane * 16;
    v16bf ahi = *(const v16bf*)(Ahi + aoff);
    v16bf alo = *(const v16bf*)(Alo + aoff);

#pragma unroll
    for (int ns = 0; ns < 4; ++ns) {
      int nbL = waveN * 4 + ns;
      v16bf bhi = *(const v16bf*)(ldsB + bufOff + nbL * 1024 + lane * 32);
      v16bf blo = *(const v16bf*)(ldsB + bufOff + 8192 + nbL * 1024 + lane * 32);
      acc[ns] = __builtin_amdgcn_wmma_f32_16x16x32_bf16(
          false, ahi, false, bhi, (short)0, acc[ns], false, false);
      acc[ns] = __builtin_amdgcn_wmma_f32_16x16x32_bf16(
          false, ahi, false, blo, (short)0, acc[ns], false, false);
      acc[ns] = __builtin_amdgcn_wmma_f32_16x16x32_bf16(
          false, alo, false, bhi, (short)0, acc[ns], false, false);
    }
    __syncthreads();
  }

  // ---- fused epilogue ----
  // C/D layout: vgpr r -> row M = r + 8*half ; col N = lane&15 within frag
  float sum = 0.f;
  const int Mstart = blockM * 64 + waveM * 16;
  if (EPI == 0) {
    float c1r[8], wr[8];
#pragma unroll
    for (int r = 0; r < 8; ++r) {
      int row = Mstart + r + 8 * hf;
      c1r[r] = c1[row];
      wr[r]  = w[row];
    }
#pragma unroll
    for (int ns = 0; ns < 4; ++ns) {
      int col = blockN * 128 + (waveN * 4 + ns) * 16 + (lane & 15);
      float nn = negnorm[col];
#pragma unroll
      for (int r = 0; r < 8; ++r) {
        float p = c1r[r] + 2.0f * acc[ns][r] - nn;
        sum += wr[r] * fmaxf(p, 0.0f);
      }
    }
  } else {
#pragma unroll
    for (int ns = 0; ns < 4; ++ns) {
      int col = blockN * 128 + (waveN * 4 + ns) * 16 + (lane & 15);
#pragma unroll
      for (int r = 0; r < 8; ++r) {
        int row = Mstart + r + 8 * hf;
        float t = acc[ns][r] - ((row == col) ? 1.0f : 0.0f);
        sum += t * t;
      }
    }
  }
  sum = wave_sum(sum);
  if (lane == 0) wsums[wid] = sum;
  __syncthreads();
  if (tid == 0) {
    float s = 0.f;
#pragma unroll
    for (int i = 0; i < 8; ++i) s += wsums[i];
    partials[blockIdx.y * gridDim.x + blockIdx.x] = s;
  }
}

// ---------------------------------------------------------------------------
// Focal triplet term: one wave per row. 8-way shfl argmin over K=512,
// then 8 gathered squared distances against F rows.
// ---------------------------------------------------------------------------
__global__ void jt_kernel(const float* __restrict__ g,
                          const float* __restrict__ vhat,
                          const float* __restrict__ F,
                          const float* __restrict__ td,
                          const float* __restrict__ w,
                          float* __restrict__ JtPart) {
  __shared__ float s8[8];
  int wid  = threadIdx.x >> 5;
  int lane = threadIdx.x & 31;
  int b = blockIdx.x * 8 + wid;

  float gv[16];
  int   gi[16];
#pragma unroll
  for (int i = 0; i < 16; ++i) {
    int idx = lane + i * 32;
    gv[i] = g[(size_t)b * Kk + idx];
    gi[i] = idx;
  }
  float selv[Tt];
  int   seli[Tt];
#pragma unroll
  for (int t = 0; t < Tt; ++t) {
    float mv = 3.402823466e38f;
    int   mi = 1 << 30;
#pragma unroll
    for (int i = 0; i < 16; ++i) {
      if (gv[i] < mv || (gv[i] == mv && gi[i] < mi)) { mv = gv[i]; mi = gi[i]; }
    }
#pragma unroll
    for (int off = 16; off > 0; off >>= 1) {
      float ov = __shfl_xor(mv, off, 32);
      int   oi = __shfl_xor(mi, off, 32);
      if (ov < mv || (ov == mv && oi < mi)) { mv = ov; mi = oi; }
    }
    selv[t] = mv;
    seli[t] = mi;
#pragma unroll
    for (int i = 0; i < 16; ++i)
      if (gi[i] == mi) gv[i] = 3.402823466e38f;
  }
  float gsum = 0.f;
#pragma unroll
  for (int t = 0; t < Tt; ++t) gsum += selv[t];
  float inv = 1.0f / (gsum + EPS);

  const float4* ph = (const float4*)(vhat + (size_t)b * Dd);
  float4 h0 = ph[lane * 2], h1 = ph[lane * 2 + 1];
  float tdb = td[b];

  float cursum = 0.f;
#pragma unroll
  for (int t = 0; t < Tt; ++t) {
    const float4* Fp = (const float4*)(F + (size_t)seli[t] * Dd);
    float4 f0 = Fp[lane * 2], f1 = Fp[lane * 2 + 1];
    float d0 = h0.x - f0.x, d1 = h0.y - f0.y, d2 = h0.z - f0.z, d3 = h0.w - f0.w;
    float d4 = h1.x - f1.x, d5 = h1.y - f1.y, d6 = h1.z - f1.z, d7 = h1.w - f1.w;
    float dl = d0*d0 + d1*d1 + d2*d2 + d3*d3 + d4*d4 + d5*d5 + d6*d6 + d7*d7;
    float dist = wave_sum(dl);
    float gt = selv[t] * inv;
    float mt = (1.0f - gt) * (1.0f - gt);   // M_MARGIN = 1
    cursum += fmaxf(mt + tdb - dist, 0.0f);
  }
  if (lane == 0) s8[wid] = cursum * w[b];
  __syncthreads();
  if (threadIdx.x == 0) {
    float x = 0.f;
#pragma unroll
    for (int i = 0; i < 8; ++i) x += s8[i];
    JtPart[blockIdx.x] = x;
  }
}

// ---------------------------------------------------------------------------
// Final deterministic reduction + scalar combine.
// ---------------------------------------------------------------------------
__device__ float block_sum256(float x, float* red) {
  int tid = threadIdx.x;
  red[tid] = x;
  __syncthreads();
  for (int s = 128; s > 0; s >>= 1) {
    if (tid < s) red[tid] += red[tid + s];
    __syncthreads();
  }
  float r = red[0];
  __syncthreads();
  return r;
}

__global__ void final_kernel(const unsigned char* __restrict__ mask,
                             const float* __restrict__ JuPart, int nJu,
                             const float* __restrict__ JtPart, int nJt,
                             const float* __restrict__ OrthoPart, int nOr,
                             float* __restrict__ out) {
  __shared__ float red[256];
  int tid = threadIdx.x;
  float msum = 0.f, ju = 0.f, jt = 0.f, oo = 0.f;
  for (int i = tid; i < Bq;  i += 256) msum += mask[i] ? 1.0f : 0.0f;
  for (int i = tid; i < nJu; i += 256) ju += JuPart[i];
  for (int i = tid; i < nJt; i += 256) jt += JtPart[i];
  for (int i = tid; i < nOr; i += 256) oo += OrthoPart[i];
  msum = block_sum256(msum, red);
  ju   = block_sum256(ju, red);
  jt   = block_sum256(jt, red);
  oo   = block_sum256(oo, red);
  if (tid == 0) {
    float Ju = (msum > 0.f) ? ju / ((float)Nn * msum) : 0.0f;
    float Jt = (msum > 0.f) ? jt / msum : 0.0f;
    out[0] = Ju + Jt + LAMBDA_ORTHO * oo;
  }
}

// ---------------------------------------------------------------------------
extern "C" void kernel_launch(void* const* d_in, const int* in_sizes, int n_in,
                              void* d_out, int out_size, void* d_ws,
                              size_t ws_size, hipStream_t stream) {
  const float*         v    = (const float*)d_in[0];
  const float*         vhat = (const float*)d_in[1];
  const float*         g    = (const float*)d_in[2];
  const float*         F    = (const float*)d_in[3];
  const float*         neg  = (const float*)d_in[4];
  const unsigned char* mask = (const unsigned char*)d_in[5];
  float* out = (float*)d_out;

  // ---- workspace bump allocator (256 B aligned) ----
  char* p = (char*)d_ws;
  auto take = [&](size_t bytes) {
    char* r = p;
    p += (bytes + 255) & ~(size_t)255;
    return r;
  };
  __bf16* Ahi  = (__bf16*)take((size_t)Bq * Dd * 2);
  __bf16* Alo  = (__bf16*)take((size_t)Bq * Dd * 2);
  __bf16* Bhi  = (__bf16*)take((size_t)Nn * Dd * 2);
  __bf16* Blo  = (__bf16*)take((size_t)Nn * Dd * 2);
  __bf16* FAhi = (__bf16*)take((size_t)Kk * Dd * 2);
  __bf16* FAlo = (__bf16*)take((size_t)Kk * Dd * 2);
  __bf16* FBhi = (__bf16*)take((size_t)Kk * Dd * 2);
  __bf16* FBlo = (__bf16*)take((size_t)Kk * Dd * 2);
  float* c1  = (float*)take((size_t)Bq * 4);
  float* td  = (float*)take((size_t)Bq * 4);
  float* w   = (float*)take((size_t)Bq * 4);
  float* nn  = (float*)take((size_t)Nn * 4);
  const int nJu = (Nn / 128) * (Bq / 64);   // 16 * 128 = 2048
  const int nJt = Bq / 8;                   // 1024
  const int nOr = (Kk / 128) * (Kk / 64);   // 4 * 8 = 32
  float* JuPart = (float*)take((size_t)nJu * 4);
  float* JtPart = (float*)take((size_t)nJt * 4);
  float* OrPart = (float*)take((size_t)nOr * 4);

  // ---- stats ----
  row_stats_kernel<<<Bq / 8, 256, 0, stream>>>(v, vhat, mask, c1, td, w);
  rownorm_kernel<<<Nn / 8, 256, 0, stream>>>(neg, nn, Nn);

  // ---- pack to WMMA fragment layouts (hi/lo bf16 split) ----
  pack_a_kernel<<<(Bq / 16) * KB / 8, 256, 0, stream>>>(vhat, Ahi, Alo, Bq);
  pack_b_kernel<<<(Nn / 16) * KB / 8, 256, 0, stream>>>(neg, Bhi, Blo, Nn);
  pack_a_kernel<<<(Kk / 16) * KB / 8, 256, 0, stream>>>(F, FAhi, FAlo, Kk);
  pack_b_kernel<<<(Kk / 16) * KB / 8, 256, 0, stream>>>(F, FBhi, FBlo, Kk);

  // ---- fused WMMA GEMMs ----
  gemm_kernel<0><<<dim3(Nn / 128, Bq / 64), 256, 0, stream>>>(
      Ahi, Alo, Bhi, Blo, c1, w, nn, JuPart);
  gemm_kernel<1><<<dim3(Kk / 128, Kk / 64), 256, 0, stream>>>(
      FAhi, FAlo, FBhi, FBlo, nullptr, nullptr, nullptr, OrPart);

  // ---- focal triplet ----
  jt_kernel<<<Bq / 8, 256, 0, stream>>>(g, vhat, F, td, w, JtPart);

  // ---- combine ----
  final_kernel<<<1, 256, 0, stream>>>(mask, JuPart, nJu, JtPart, nJt, OrPart,
                                      nOr, out);
}
